// ANEncoder_37580963840020
// MI455X (gfx1250) — compile-verified, matrix operand order checked by reading, not run on hardware
//
#include <hip/hip_runtime.h>

typedef __attribute__((ext_vector_type(16))) _Float16 v16h;
typedef __attribute__((ext_vector_type(8)))  _Float16 v8h;
typedef __attribute__((ext_vector_type(8)))  float    v8f;

// ---------------------------------------------------------------- utilities

__device__ __forceinline__ void atomicMaxF(float* addr, float val) {
    int* ia = (int*)addr;
    int old = *ia;
    while (__int_as_float(old) < val) {
        int assumed = old;
        old = atomicCAS(ia, assumed, __float_as_int(val));
        if (old == assumed) break;
    }
}

__global__ void k_build_edges(const int* __restrict__ ei, int* __restrict__ src,
                              int* __restrict__ dst, int E, int Nn) {
    int i = blockIdx.x * blockDim.x + threadIdx.x;
    if (i >= E + Nn) return;
    if (i < E) { src[i] = ei[i]; dst[i] = ei[E + i]; }
    else       { src[i] = i - E; dst[i] = i - E; }
}

__global__ void k_fill_f32(float* __restrict__ p, float v, long n) {
    long i = (long)blockIdx.x * blockDim.x + threadIdx.x;
    if (i < n) p[i] = v;
}

// cast fp32 node features -> f16, zero-pad rows to Mpad and cols to Kp
__global__ void k_cast_pad_x(const float* __restrict__ x, _Float16* __restrict__ out,
                             int Nn, int fi, int Kp, long total) {
    long i = (long)blockIdx.x * blockDim.x + threadIdx.x;
    if (i >= total) return;
    int r = (int)(i / Kp), k = (int)(i % Kp);
    float v = (r < Nn && k < fi) ? x[(size_t)r * fi + k] : 0.f;
    out[i] = (_Float16)v;
}

// W[fi, H*C] fp32 row-major  ->  Wt[h][Cp][Kp] f16 (B^T layout, zero padded)
__global__ __launch_bounds__(256)
void k_transpose_w(const float* __restrict__ W, _Float16* __restrict__ Wt,
                   int fi, int C, int Cp, int Kp, int HC) {
    __shared__ _Float16 t[32][33];
    int h = blockIdx.z;
    int c0 = blockIdx.x * 32, k0 = blockIdx.y * 32;
#pragma unroll
    for (int i = 0; i < 4; ++i) {
        int c = c0 + threadIdx.x;
        int k = k0 + threadIdx.y + i * 8;
        float v = (c < C && k < fi) ? W[(size_t)k * HC + (size_t)h * C + c] : 0.f;
        t[threadIdx.y + i * 8][threadIdx.x] = (_Float16)v;
    }
    __syncthreads();
#pragma unroll
    for (int i = 0; i < 4; ++i) {
        int k = k0 + threadIdx.x;
        int c = c0 + threadIdx.y + i * 8;
        Wt[((size_t)h * Cp + c) * Kp + k] = t[threadIdx.x][threadIdx.y + i * 8];
    }
}

// ---------------------------------------------------------------- WMMA GEMM
// Per-lane fragment load for 16-bit 16x32 A (and B^T, identical pattern):
// lanes 0-15 hold rows 0-15 with K = [k0..k0+7] + [k0+16..k0+23],
// lanes 16-31 hold rows 0-15 with K = [k0+8..k0+15] + [k0+24..k0+31].
__device__ __forceinline__ v16h load_frag16(const _Float16* __restrict__ base,
                                            int ld, int row, int k0, int lane) {
    const _Float16* q = base + (size_t)(row + (lane & 15)) * ld + k0 + ((lane >> 4) << 3);
    union { v16h v; v8h h[2]; } u;
    u.h[0] = *(const v8h*)(q);
    u.h[1] = *(const v8h*)(q + 16);
    return u.v;
}

__device__ __forceinline__ void store_ctile(const v8f& c, _Float16* __restrict__ O,
                                            int Nn, int Cp, int C,
                                            const float* __restrict__ bh,
                                            int mBase, int col, int rOff) {
    float b = (col < C) ? bh[col] : 0.f;
#pragma unroll
    for (int i = 0; i < 8; ++i) {
        int r = mBase + rOff + i;
        if (r < Nn) O[(size_t)r * Cp + col] = (_Float16)(c[i] + b);
    }
}

// C = A[Mpad,Kp] (f16) * Bt[h][Cp,Kp]^T + bias  ->  f16 Out[hz][Nn,Cp]
// block = 128 threads = 4 waves, each wave does a 32x32 tile (2x2 WMMA tiles).
// Software-pipelined: K-step k+32 fragment loads issued before the WMMAs of
// step k, so the v_wmma waits land one stage behind the in-flight b128 loads.
__global__ __launch_bounds__(128)
void k_gemm_f16(const _Float16* __restrict__ A, const _Float16* __restrict__ Bt,
                const float* __restrict__ bias, _Float16* __restrict__ Out,
                int Nn, int Kp, int Cp, int C, int chunk, int CHn) {
    const int lane = threadIdx.x & 31;
    const int wid  = threadIdx.x >> 5;
    const int hz   = blockIdx.z;
    const int h    = chunk * CHn + hz;
    const _Float16* B = Bt + (size_t)h * Cp * Kp;
    _Float16* O = Out + (size_t)hz * Nn * Cp;

    const int mBase = blockIdx.x * 64 + (wid >> 1) * 32;
    const int nBase = blockIdx.y * 64 + (wid & 1) * 32;

    v8f c00 = {}, c01 = {}, c10 = {}, c11 = {};
    v16h a0 = load_frag16(A, Kp, mBase,      0, lane);
    v16h a1 = load_frag16(A, Kp, mBase + 16, 0, lane);
    v16h b0 = load_frag16(B, Kp, nBase,      0, lane);
    v16h b1 = load_frag16(B, Kp, nBase + 16, 0, lane);
    for (int k0 = 32; k0 < Kp; k0 += 32) {
        v16h na0 = load_frag16(A, Kp, mBase,      k0, lane);
        v16h na1 = load_frag16(A, Kp, mBase + 16, k0, lane);
        v16h nb0 = load_frag16(B, Kp, nBase,      k0, lane);
        v16h nb1 = load_frag16(B, Kp, nBase + 16, k0, lane);
        c00 = __builtin_amdgcn_wmma_f32_16x16x32_f16(false, a0, false, b0, (short)0, c00, false, false);
        c01 = __builtin_amdgcn_wmma_f32_16x16x32_f16(false, a0, false, b1, (short)0, c01, false, false);
        c10 = __builtin_amdgcn_wmma_f32_16x16x32_f16(false, a1, false, b0, (short)0, c10, false, false);
        c11 = __builtin_amdgcn_wmma_f32_16x16x32_f16(false, a1, false, b1, (short)0, c11, false, false);
        a0 = na0; a1 = na1; b0 = nb0; b1 = nb1;
    }
    c00 = __builtin_amdgcn_wmma_f32_16x16x32_f16(false, a0, false, b0, (short)0, c00, false, false);
    c01 = __builtin_amdgcn_wmma_f32_16x16x32_f16(false, a0, false, b1, (short)0, c01, false, false);
    c10 = __builtin_amdgcn_wmma_f32_16x16x32_f16(false, a1, false, b0, (short)0, c10, false, false);
    c11 = __builtin_amdgcn_wmma_f32_16x16x32_f16(false, a1, false, b1, (short)0, c11, false, false);

    const float* bh = bias + (size_t)h * C;
    const int rOff = (lane >> 4) * 8;
    const int col  = nBase + (lane & 15);
    store_ctile(c00, O, Nn, Cp, C, bh, mBase,      col,      rOff);
    store_ctile(c01, O, Nn, Cp, C, bh, mBase,      col + 16, rOff);
    store_ctile(c10, O, Nn, Cp, C, bh, mBase + 16, col,      rOff);
    store_ctile(c11, O, Nn, Cp, C, bh, mBase + 16, col + 16, rOff);
}

// ---------------------------------------------------------------- edge phase

// one wave per (edge, local head): e = sum_c lrelu(xl[src]+xr[dst])*att ; atomic max per dst.
// XL/XR pad columns are exact zeros, so we iterate the padded Cp with unguarded
// b128 vector loads; att is staged zero-padded into LDS (all waves share head h).
__global__ __launch_bounds__(256)
void k_edge_logits(const _Float16* __restrict__ XL, const _Float16* __restrict__ XR,
                   const float* __restrict__ att, const int* __restrict__ src,
                   const int* __restrict__ dst, float* __restrict__ elog,
                   float* __restrict__ mrow, int E2, int Nn, int C, int Cp,
                   int chunk, int CHn) {
    __shared__ float s_att[1088];
    const int hz = blockIdx.y;
    const int h  = chunk * CHn + hz;
    const float* a = att + (size_t)h * C;
    for (int i = threadIdx.x; i < Cp; i += 256)
        s_att[i] = (i < C) ? a[i] : 0.f;
    __syncthreads();

    const int lane = threadIdx.x & 31;
    const int w = blockIdx.x * 8 + (threadIdx.x >> 5);
    if (w >= E2) return;
    const int s = src[w], d = dst[w];
    const _Float16* xl = XL + ((size_t)hz * Nn + s) * Cp;
    const _Float16* xr = XR + ((size_t)hz * Nn + d) * Cp;
    float sum = 0.f;
    for (int c0 = lane * 8; c0 < Cp; c0 += 256) {
        v8h xlv = *(const v8h*)(xl + c0);
        v8h xrv = *(const v8h*)(xr + c0);
#pragma unroll
        for (int j = 0; j < 8; ++j) {
            float v = (float)xlv[j] + (float)xrv[j];
            v = v > 0.f ? v : 0.2f * v;
            sum += v * s_att[c0 + j];
        }
    }
#pragma unroll
    for (int off = 16; off; off >>= 1) sum += __shfl_xor(sum, off, 32);
    if (lane == 0) {
        elog[(size_t)hz * E2 + w] = sum;
        atomicMaxF(&mrow[(size_t)hz * Nn + d], sum);
    }
}

__global__ void k_edge_exp(const float* __restrict__ elog, const float* __restrict__ mrow,
                           const int* __restrict__ dst, float* __restrict__ p,
                           float* __restrict__ den, int E2, int Nn) {
    int e = blockIdx.x * blockDim.x + threadIdx.x;
    if (e >= E2) return;
    int hz = blockIdx.y;
    int d = dst[e];
    float v = expf(elog[(size_t)hz * E2 + e] - mrow[(size_t)hz * Nn + d]);
    p[(size_t)hz * E2 + e] = v;
    atomicAdd(&den[(size_t)hz * Nn + d], v);
}

// one wave per (edge, local head): acc[dst] += (p/den) * xl[src] / H
// b128 vector loads of xl; atomics guarded to the unpadded C columns.
__global__ __launch_bounds__(256)
void k_edge_aggregate(const _Float16* __restrict__ XL, const float* __restrict__ p,
                      const float* __restrict__ den, const int* __restrict__ src,
                      const int* __restrict__ dst, float* __restrict__ acc,
                      int E2, int Nn, int C, int Cp) {
    const int lane = threadIdx.x & 31;
    const int w = blockIdx.x * 8 + (threadIdx.x >> 5);
    if (w >= E2) return;
    const int hz = blockIdx.y;
    const int s = src[w], d = dst[w];
    const float alpha = p[(size_t)hz * E2 + w] / den[(size_t)hz * Nn + d] * (1.0f / 32.0f);
    const _Float16* xl = XL + ((size_t)hz * Nn + s) * Cp;
    float* ao = acc + (size_t)d * C;
    for (int c0 = lane * 8; c0 < Cp; c0 += 256) {
        v8h xlv = *(const v8h*)(xl + c0);
#pragma unroll
        for (int j = 0; j < 8; ++j) {
            int c = c0 + j;
            if (c < C) atomicAdd(&ao[c], alpha * (float)xlv[j]);
        }
    }
}

// out = acc + bias ; leaky_relu(0.01) + cast to f16 for next layer, or fp32 final
__global__ void k_finalize(const float* __restrict__ acc, const float* __restrict__ bias,
                           _Float16* __restrict__ xnext, float* __restrict__ dout,
                           int Nn, int C, int Kpn, int last, float slope) {
    long i = (long)blockIdx.x * blockDim.x + threadIdx.x;
    if (i >= (long)Nn * C) return;
    int r = (int)(i / C), c = (int)(i % C);
    float v = acc[i] + bias[c];
    if (!last) {
        v = v > 0.f ? v : slope * v;
        xnext[(size_t)r * Kpn + c] = (_Float16)v;
    } else {
        dout[i] = v;
    }
}

// ---------------------------------------------------------------- host driver

extern "C" void kernel_launch(void* const* d_in, const int* in_sizes, int n_in,
                              void* d_out, int out_size, void* d_ws, size_t ws_size,
                              hipStream_t stream) {
    (void)in_sizes; (void)n_in; (void)out_size; (void)ws_size;
    const int Nn = 10000, E = 160000, H = 32, CH = 4;
    const int E2 = E + Nn;            // edges + self loops
    const int Mpad = 10048;           // 157 * 64
    const int NCHUNK = H / CH;

    const float* x0 = (const float*)d_in[0];
    const int*   ei = (const int*)d_in[1];

    // workspace carve-out (256B aligned bump allocator), ~304 MB total
    char* ws = (char*)d_ws;
    size_t off = 0;
    auto alloc = [&](size_t bytes) -> void* {
        void* p = ws + off;
        off += (bytes + 255) & ~(size_t)255;
        return p;
    };
    int*      srcA = (int*)alloc((size_t)E2 * 4);
    int*      dstA = (int*)alloc((size_t)E2 * 4);
    _Float16* xf16 = (_Float16*)alloc((size_t)Mpad * 512 * 2);
    _Float16* WtL  = (_Float16*)alloc((size_t)H * 1088 * 512 * 2);
    _Float16* WtR  = (_Float16*)alloc((size_t)H * 1088 * 512 * 2);
    _Float16* XLb  = (_Float16*)alloc((size_t)CH * Nn * 1088 * 2);
    _Float16* XRb  = (_Float16*)alloc((size_t)CH * Nn * 1088 * 2);
    float*    elog = (float*)alloc((size_t)CH * E2 * 4);
    float*    pbuf = (float*)alloc((size_t)CH * E2 * 4);
    float*    mrow = (float*)alloc((size_t)CH * Nn * 4);
    float*    den  = (float*)alloc((size_t)CH * Nn * 4);
    float*    acc  = (float*)alloc((size_t)Nn * 1028 * 4);

    k_build_edges<<<(E2 + 255) / 256, 256, 0, stream>>>(ei, srcA, dstA, E, Nn);
    {
        long total = (long)Mpad * 32;
        k_cast_pad_x<<<(int)((total + 255) / 256), 256, 0, stream>>>(x0, xf16, Nn, 4, 32, total);
    }

    const int fis[3] = {4, 128, 512};
    const int fos[3] = {128, 512, 1028};
    const int Kps[3] = {32, 128, 512};     // padded K (mult of 32)
    const int Cps[3] = {128, 512, 1088};   // padded per-head C (mult of 64)
    const int KpN[3] = {128, 512, 0};      // next layer's padded K

    for (int l = 0; l < 3; ++l) {
        const float* Wl   = (const float*)d_in[2 + l * 6 + 0];
        const float* bl   = (const float*)d_in[2 + l * 6 + 1];
        const float* Wr   = (const float*)d_in[2 + l * 6 + 2];
        const float* br   = (const float*)d_in[2 + l * 6 + 3];
        const float* att  = (const float*)d_in[2 + l * 6 + 4];
        const float* bias = (const float*)d_in[2 + l * 6 + 5];
        const int fi = fis[l], C = fos[l], Kp = Kps[l], Cp = Cps[l], HC = H * C;

        dim3 tg(Cp / 32, Kp / 32, H), tb(32, 8);
        k_transpose_w<<<tg, tb, 0, stream>>>(Wl, WtL, fi, C, Cp, Kp, HC);
        k_transpose_w<<<tg, tb, 0, stream>>>(Wr, WtR, fi, C, Cp, Kp, HC);

        {
            long n = (long)Nn * C;
            k_fill_f32<<<(int)((n + 255) / 256), 256, 0, stream>>>(acc, 0.f, n);
        }

        for (int chunk = 0; chunk < NCHUNK; ++chunk) {
            long nm = (long)CH * Nn;
            k_fill_f32<<<(int)((nm + 255) / 256), 256, 0, stream>>>(mrow, -1e30f, nm);
            k_fill_f32<<<(int)((nm + 255) / 256), 256, 0, stream>>>(den, 0.f, nm);

            dim3 gg(Mpad / 64, Cp / 64, CH);
            k_gemm_f16<<<gg, 128, 0, stream>>>(xf16, WtL, bl, XLb, Nn, Kp, Cp, C, chunk, CH);
            k_gemm_f16<<<gg, 128, 0, stream>>>(xf16, WtR, br, XRb, Nn, Kp, Cp, C, chunk, CH);

            dim3 ge((E2 + 7) / 8, CH);
            k_edge_logits<<<ge, 256, 0, stream>>>(XLb, XRb, att, srcA, dstA, elog, mrow,
                                                  E2, Nn, C, Cp, chunk, CH);
            dim3 gx((E2 + 255) / 256, CH);
            k_edge_exp<<<gx, 256, 0, stream>>>(elog, mrow, dstA, pbuf, den, E2, Nn);
            k_edge_aggregate<<<ge, 256, 0, stream>>>(XLb, pbuf, den, srcA, dstA, acc,
                                                     E2, Nn, C, Cp);
        }

        long nf = (long)Nn * C;
        if (l < 2) {
            long nz = (long)Mpad * KpN[l] / 2;  // zero f16 buffer via f32 fills
            k_fill_f32<<<(int)((nz + 255) / 256), 256, 0, stream>>>((float*)xf16, 0.f, nz);
            k_finalize<<<(int)((nf + 255) / 256), 256, 0, stream>>>(acc, bias, xf16, nullptr,
                                                                    Nn, C, KpN[l], 0, 0.01f);
        } else {
            k_finalize<<<(int)((nf + 255) / 256), 256, 0, stream>>>(acc, bias, nullptr,
                                                                    (float*)d_out, Nn, C, 0, 1, 0.f);
        }
    }
}
